// GCNLayer_47321949667967
// MI455X (gfx1250) — matrix operand order repaired and checked by reading, not compiled, not run on hardware
//
#include <hip/hip_runtime.h>

#define IN_DIM 256
#define OUT_DIM 256
#define USE_ASYNC_LDS 1

typedef __attribute__((ext_vector_type(16))) __bf16 v16bf;
typedef __attribute__((ext_vector_type(8)))  float v8f;

__device__ __forceinline__ unsigned short f2bf(float f) {
  unsigned u = __builtin_bit_cast(unsigned, f);
  u += 0x7FFFu + ((u >> 16) & 1u);          // round-to-nearest-even
  return (unsigned short)(u >> 16);
}
__device__ __forceinline__ unsigned pack2bf(float a, float b) {
  return (unsigned)f2bf(a) | ((unsigned)f2bf(b) << 16);
}

// ---------------------------------------------------------------------------
// Pre-pass 1: x (f32) -> xb (bf16, packed dwords), streaming.
// ---------------------------------------------------------------------------
__global__ void convert_x_bf16(const float* __restrict__ x,
                               unsigned* __restrict__ xb, int total8)
{
  int i = blockIdx.x * blockDim.x + threadIdx.x;
  int s = gridDim.x * blockDim.x;
  for (; i < total8; i += s) {
    float4 a = ((const float4*)x)[2 * i];
    float4 b = ((const float4*)x)[2 * i + 1];
    uint4 o;
    o.x = pack2bf(a.x, a.y); o.y = pack2bf(a.z, a.w);
    o.z = pack2bf(b.x, b.y); o.w = pack2bf(b.z, b.w);
    ((uint4*)xb)[i] = o;
  }
}

// ---------------------------------------------------------------------------
// Pre-pass 2: W (f32, [K][N]) -> Wt (bf16 packed, transposed [col][K]).
// Tiny (128 KB out); one block.
// ---------------------------------------------------------------------------
__global__ void __launch_bounds__(256)
convert_wt_bf16(const float* __restrict__ W, unsigned* __restrict__ Wt)
{
  const int col = threadIdx.x;
  for (int k = 0; k < IN_DIM; k += 2) {
    float w0 = W[(size_t)k * OUT_DIM + col];
    float w1 = W[(size_t)(k + 1) * OUT_DIM + col];
    Wt[col * (IN_DIM / 2) + (k >> 1)] = pack2bf(w0, w1);
  }
}

// ---------------------------------------------------------------------------
// Kernel: h = x @ W via bf16 WMMA (f32 accumulate).
// Block: 256 threads = 8 waves; wave -> 16 rows x 128 cols.
// W slice staged to LDS via global_load_async_to_lds_b128 (ASYNCcnt path).
// LDS layout: [col][36 dwords] (32 dwords = 64 K bf16 + 4 pad), 16B aligned.
// ---------------------------------------------------------------------------
__global__ void __launch_bounds__(256)
gemm_bf16_wmma(const unsigned* __restrict__ xb, const unsigned* __restrict__ Wt,
               float* __restrict__ h, int N)
{
  __shared__ unsigned wlds[OUT_DIM * 36];   // 36,864 B

  const int tid    = threadIdx.x;
  const int lane   = tid & 31;
  const int wid    = tid >> 5;
  const int waveM  = wid >> 1;              // 0..3
  const int waveN  = wid & 1;               // 0..1
  const int rowBase = blockIdx.x * 64 + waveM * 16;
  const int colBase = waveN * 128;
  const int mrow   = lane & 15;
  const bool hiHalf = lane >= 16;

  v8f acc[8];
  #pragma unroll
  for (int t = 0; t < 8; ++t)
    acc[t] = (v8f){0.f,0.f,0.f,0.f,0.f,0.f,0.f,0.f};

  const int r = min(rowBase + mrow, N - 1); // clamp: padded rows never stored
  const unsigned* xrow = xb + (size_t)r * (IN_DIM / 2);   // 128 dwords / row
  const int koffA = hiHalf ? 8 : 0;

#if USE_ASYNC_LDS
  const unsigned ldsBase = (unsigned)(uintptr_t)&wlds[0]; // LDS byte offset
  const unsigned long long wtBase = (unsigned long long)Wt;
#endif

  for (int ks = 0; ks < 4; ++ks) {          // 4 slices of 64 K-values
    __syncthreads();
    // Stage Wt[col][ks*64 .. +63] (128 B/col) into LDS row col*144 bytes.
    {
      const unsigned col = (unsigned)tid;
#if USE_ASYNC_LDS
      const unsigned gbase = (col * IN_DIM + (unsigned)ks * 64) * 2; // bytes
      #pragma unroll
      for (int i = 0; i < 8; ++i) {
        unsigned laddr = ldsBase + col * 144u + (unsigned)i * 16u;
        unsigned gaddr = gbase + (unsigned)i * 16u;
        asm volatile("global_load_async_to_lds_b128 %0, %1, %2"
                     :: "v"(laddr), "v"(gaddr), "s"(wtBase) : "memory");
      }
#else
      const uint4* src = (const uint4*)(Wt + (size_t)col * (IN_DIM / 2) + ks * 32);
      uint4* dst = (uint4*)((char*)wlds + (size_t)col * 144);
      #pragma unroll
      for (int i = 0; i < 8; ++i) dst[i] = src[i];
#endif
    }
#if USE_ASYNC_LDS
    asm volatile("s_wait_asynccnt 0" ::: "memory");
#endif
    __syncthreads();

    #pragma unroll
    for (int kb = 0; kb < 2; ++kb) {        // two K=32 WMMA steps per slice
      const int kg = ks * 64 + kb * 32;
      // A fragment: row mrow, bf16 K = {koffA..+7, 16+koffA..+7}
      const int dw = (kg + koffA) >> 1;     // dword offset, multiple of 4
      union { v16bf v; uint4 q[2]; } A;
      A.q[0] = *(const uint4*)(xrow + dw);
      A.q[1] = *(const uint4*)(xrow + dw + 8);

      const int dwB = kb * 16 + (hiHalf ? 4 : 0);
      #pragma unroll
      for (int t = 0; t < 8; ++t) {
        const int col = colBase + t * 16 + mrow;
        const unsigned* bp = wlds + col * 36 + dwB;
        union { v16bf v; uint4 q[2]; } B;
        B.q[0] = *(const uint4*)(bp);
        B.q[1] = *(const uint4*)(bp + 8);
        acc[t] = __builtin_amdgcn_wmma_f32_16x16x32_bf16(
            false, A.v, false, B.v, (short)0, acc[t], false, false);
      }
    }
  }

  // C/D layout: VGPR i -> M = i + (hiHalf?8:0), N = lane%16
  const int srow0 = rowBase + (hiHalf ? 8 : 0);
  #pragma unroll
  for (int t = 0; t < 8; ++t) {
    const int col = colBase + t * 16 + mrow;
    #pragma unroll
    for (int i = 0; i < 8; ++i) {
      int rr = srow0 + i;
      if (rr < N) h[(size_t)rr * OUT_DIM + col] = acc[t][i];
    }
  }
}

// ---------------------------------------------------------------------------
// CSR construction: zero -> count -> hierarchical scan (3 phases) -> scatter
// ---------------------------------------------------------------------------
__global__ void zero_int(int* __restrict__ p, int n) {
  int i = blockIdx.x * blockDim.x + threadIdx.x;
  int s = gridDim.x * blockDim.x;
  for (; i < n; i += s) p[i] = 0;
}

__global__ void count_rows(const int* __restrict__ rows, int* __restrict__ cnt, int nnz) {
  int i = blockIdx.x * blockDim.x + threadIdx.x;
  int s = gridDim.x * blockDim.x;
  for (; i < nnz; i += s) atomicAdd(&cnt[rows[i]], 1);
}

__device__ __forceinline__ int block_scan_inclusive(int* buf, int tid, int v) {
  buf[tid] = v;
  __syncthreads();
  for (int d = 1; d < 1024; d <<= 1) {
    int t = (tid >= d) ? buf[tid - d] : 0;
    __syncthreads();
    buf[tid] += t;
    __syncthreads();
  }
  return buf[tid];
}

__global__ void __launch_bounds__(1024)
scan_phaseA(const int* __restrict__ cnt, int* __restrict__ offs,
            int* __restrict__ bsum, int n)
{
  __shared__ int buf[1024];
  const int tid = threadIdx.x;
  const int i = blockIdx.x * 1024 + tid;
  int v = (i < n) ? cnt[i] : 0;
  int incl = block_scan_inclusive(buf, tid, v);
  if (i < n) offs[i] = incl - v;           // block-local exclusive
  if (tid == 1023) bsum[blockIdx.x] = incl;
}

__global__ void __launch_bounds__(1024)
scan_phaseB(const int* __restrict__ bsum, int* __restrict__ bbase,
            int* __restrict__ offs, int n, int nb)
{
  __shared__ int buf[1024];
  const int tid = threadIdx.x;
  int v = (tid < nb) ? bsum[tid] : 0;
  int incl = block_scan_inclusive(buf, tid, v);
  if (tid < nb) bbase[tid] = incl - v;     // exclusive block bases
  if (tid == nb - 1) offs[n] = incl;       // grand total
}

__global__ void __launch_bounds__(1024)
scan_phaseC(int* __restrict__ offs, int* __restrict__ cursor,
            const int* __restrict__ bbase, int n)
{
  const int i = blockIdx.x * 1024 + threadIdx.x;
  if (i < n) {
    int e = offs[i] + bbase[blockIdx.x];
    offs[i] = e;
    cursor[i] = e;
  }
}

__global__ void scatter_csr(const int* __restrict__ rows, const int* __restrict__ cols,
                            const float* __restrict__ vals, int* __restrict__ cursor,
                            int* __restrict__ csr_col, float* __restrict__ csr_val, int nnz)
{
  int i = blockIdx.x * blockDim.x + threadIdx.x;
  int s = gridDim.x * blockDim.x;
  for (; i < nnz; i += s) {
    int pos = atomicAdd(&cursor[rows[i]], 1);
    csr_col[pos] = cols[i];
    csr_val[pos] = vals[i];
  }
}

// ---------------------------------------------------------------------------
// CSR SpMM: one wave32 per row, 8 features/lane in registers.
// 32 CSR entries batched per wave: 1 coalesced load + shfl broadcast.
// ---------------------------------------------------------------------------
__global__ void __launch_bounds__(256)
spmm_csr(const float* __restrict__ h, const int* __restrict__ offs,
         const int* __restrict__ cols, const float* __restrict__ vals,
         const float* __restrict__ bias, float* __restrict__ out, int N)
{
  const int wid  = threadIdx.x >> 5;
  const int lane = threadIdx.x & 31;
  const int row  = blockIdx.x * 8 + wid;
  if (row >= N) return;
  const int c0 = lane * 8;
  float4 a0 = *(const float4*)(bias + c0);
  float4 a1 = *(const float4*)(bias + c0 + 4);
  const int start = offs[row], end = offs[row + 1];
  for (int base = start; base < end; base += 32) {
    const int m = end - base;
    int   cl = 0;
    float vl = 0.f;
    if (lane < m) { cl = cols[base + lane]; vl = vals[base + lane]; }
    const int cnt = m < 32 ? m : 32;
    for (int j = 0; j < cnt; ++j) {
      const int   c = __shfl(cl, j, 32);
      const float v = __shfl(vl, j, 32);
      const float* hr = h + (size_t)c * OUT_DIM + c0;
      const float4 h0 = *(const float4*)(hr);
      const float4 h1 = *(const float4*)(hr + 4);
      a0.x = fmaf(v, h0.x, a0.x); a0.y = fmaf(v, h0.y, a0.y);
      a0.z = fmaf(v, h0.z, a0.z); a0.w = fmaf(v, h0.w, a0.w);
      a1.x = fmaf(v, h1.x, a1.x); a1.y = fmaf(v, h1.y, a1.y);
      a1.z = fmaf(v, h1.z, a1.z); a1.w = fmaf(v, h1.w, a1.w);
    }
  }
  float* orow = out + (size_t)row * OUT_DIM + c0;
  *(float4*)(orow)     = a0;
  *(float4*)(orow + 4) = a1;
}

// ---------------------------------------------------------------------------
// BatchNorm (training stats over nodes) + ReLU
// ---------------------------------------------------------------------------
__global__ void __launch_bounds__(256)
bn_partial(const float* __restrict__ out, float* __restrict__ psum,
           float* __restrict__ psq, int N)
{
  const int c  = threadIdx.x;           // column = lane -> coalesced
  const int r0 = blockIdx.x * 256;
  const int r1 = min(r0 + 256, N);
  float s = 0.f, q = 0.f;
  for (int r = r0; r < r1; ++r) {
    float v = out[(size_t)r * OUT_DIM + c];
    s += v; q += v * v;
  }
  psum[blockIdx.x * OUT_DIM + c] = s;
  psq [blockIdx.x * OUT_DIM + c] = q;
}

__global__ void __launch_bounds__(256)
bn_finalize(const float* __restrict__ psum, const float* __restrict__ psq,
            const float* __restrict__ gamma, const float* __restrict__ beta,
            float* __restrict__ sc, float* __restrict__ sh, int N, int nblk)
{
  const int c = threadIdx.x;
  float s = 0.f, q = 0.f;
  for (int b = 0; b < nblk; ++b) {
    s += psum[b * OUT_DIM + c];
    q += psq [b * OUT_DIM + c];
  }
  const float invN = 1.0f / (float)N;
  const float mean = s * invN;
  float var = q * invN - mean * mean;
  var = fmaxf(var, 0.f);
  const float scale = gamma[c] * rsqrtf(var + 1e-5f);
  sc[c] = scale;
  sh[c] = beta[c] - mean * scale;
}

__global__ void bn_apply(float* __restrict__ out, const float* __restrict__ sc,
                         const float* __restrict__ sh, int total4)
{
  int i = blockIdx.x * blockDim.x + threadIdx.x;
  int s = gridDim.x * blockDim.x;
  for (; i < total4; i += s) {
    float4 v = ((float4*)out)[i];
    const int c = (i * 4) & (OUT_DIM - 1);
    v.x = fmaxf(0.f, fmaf(v.x, sc[c + 0], sh[c + 0]));
    v.y = fmaxf(0.f, fmaf(v.y, sc[c + 1], sh[c + 1]));
    v.z = fmaxf(0.f, fmaf(v.z, sc[c + 2], sh[c + 2]));
    v.w = fmaxf(0.f, fmaf(v.w, sc[c + 3], sh[c + 3]));
    ((float4*)out)[i] = v;
  }
}

// ---------------------------------------------------------------------------
extern "C" void kernel_launch(void* const* d_in, const int* in_sizes, int n_in,
                              void* d_out, int out_size, void* d_ws, size_t ws_size,
                              hipStream_t stream)
{
  const float* x     = (const float*)d_in[0];
  const int*   Aidx  = (const int*)  d_in[1];   // [2][NNZ]: rows then cols
  const float* Avals = (const float*)d_in[2];
  const float* W     = (const float*)d_in[4];
  const float* bias  = (const float*)d_in[5];
  const float* gamma = (const float*)d_in[6];
  const float* beta  = (const float*)d_in[7];
  float* out = (float*)d_out;

  const int N   = in_sizes[0] / IN_DIM;
  const int NNZ = in_sizes[2];
  const int* Arows = Aidx;
  const int* Acols = Aidx + NNZ;

  // Workspace carve-out (256B-aligned)
  char* ws = (char*)d_ws;
  size_t off = 0;
  auto carve = [&](size_t bytes) -> void* {
    off = (off + 255) & ~(size_t)255;
    void* p = ws + off;
    off += bytes;
    return p;
  };
  const int nblk = (N + 255) / 256;           // bn partial blocks
  const int nb   = (N + 1023) / 1024;         // scan blocks
  float*    h       = (float*)   carve((size_t)N * OUT_DIM * sizeof(float));
  unsigned* xbu     = (unsigned*)carve((size_t)N * (IN_DIM / 2) * sizeof(unsigned));
  unsigned* Wtu     = (unsigned*)carve((size_t)OUT_DIM * (IN_DIM / 2) * sizeof(unsigned));
  int*      counts  = (int*)     carve((size_t)(N + 1) * sizeof(int));
  int*      offs    = (int*)     carve((size_t)(N + 1) * sizeof(int));
  int*      cursor  = (int*)     carve((size_t)(N + 1) * sizeof(int));
  int*      bsum    = (int*)     carve((size_t)nb * sizeof(int));
  int*      bbase   = (int*)     carve((size_t)nb * sizeof(int));
  int*      csr_col = (int*)     carve((size_t)NNZ * sizeof(int));
  float*    csr_val = (float*)   carve((size_t)NNZ * sizeof(float));
  float*    psum    = (float*)   carve((size_t)nblk * OUT_DIM * sizeof(float));
  float*    psq     = (float*)   carve((size_t)nblk * OUT_DIM * sizeof(float));
  float*    sc      = (float*)   carve(OUT_DIM * sizeof(float));
  float*    sh      = (float*)   carve(OUT_DIM * sizeof(float));

  // 0) precision conversion pre-passes (streaming, HBM-bound)
  const int total8 = N * (IN_DIM / 8);
  convert_x_bf16 <<<(total8 + 1023) / 1024, 256, 0, stream>>>(x, xbu, total8);
  convert_wt_bf16<<<1, 256, 0, stream>>>(W, Wtu);

  // 1) build CSR
  zero_int   <<<((N + 1) + 255) / 256, 256, 0, stream>>>(counts, N + 1);
  count_rows <<<4096, 256, 0, stream>>>(Arows, counts, NNZ);
  scan_phaseA<<<nb, 1024, 0, stream>>>(counts, offs, bsum, N);
  scan_phaseB<<<1, 1024, 0, stream>>>(bsum, bbase, offs, N, nb);
  scan_phaseC<<<nb, 1024, 0, stream>>>(offs, cursor, bbase, N);
  scatter_csr<<<4096, 256, 0, stream>>>(Arows, Acols, Avals, cursor,
                                        csr_col, csr_val, NNZ);
  // 2) h = x @ W  (WMMA bf16, async-to-LDS W staging)
  gemm_bf16_wmma<<<(N + 63) / 64, 256, 0, stream>>>(xbu, Wtu, h, N);
  // 3) out = A @ h + bias
  spmm_csr<<<(N + 7) / 8, 256, 0, stream>>>(h, offs, csr_col, csr_val, bias, out, N);
  // 4) BatchNorm + ReLU (in place on out)
  bn_partial <<<nblk, 256, 0, stream>>>(out, psum, psq, N);
  bn_finalize<<<1, 256, 0, stream>>>(psum, psq, gamma, beta, sc, sh, N, nblk);
  const int total4 = N * (OUT_DIM / 4);
  bn_apply   <<<(total4 + 1023) / 1024, 256, 0, stream>>>(out, sc, sh, total4);
}
